// AttentionBlock_65197603553792
// MI455X (gfx1250) — compile-verified, matrix operand order checked by reading, not run on hardware
//
#include <hip/hip_runtime.h>
#include <hip/hip_bf16.h>
#include <cstddef>
#include <cstdint>

// ---------------------------------------------------------------------------
// Types for CDNA5 WMMA (wave32): v_wmma_f32_16x16x32_f16
// ---------------------------------------------------------------------------
typedef __attribute__((ext_vector_type(16))) _Float16 v16h;
typedef __attribute__((ext_vector_type(8)))  _Float16 h8;
typedef __attribute__((ext_vector_type(8)))  float    v8f;

#define WMMA_F16(a, b, c) \
  __builtin_amdgcn_wmma_f32_16x16x32_f16(false, (a), false, (b), (short)0, (c), false, false)

// CDNA5 async global->LDS copy (16B per lane), tracked with ASYNCcnt.
// ldsoff: 32-bit LDS byte offset; gaddr: 64-bit global address.
__device__ __forceinline__ void async_copy_b128(unsigned ldsoff, const void* gaddr) {
  asm volatile("global_load_async_to_lds_b128 %0, %1, off"
               :: "v"(ldsoff), "v"((unsigned long long)(uintptr_t)gaddr)
               : "memory");
}
__device__ __forceinline__ void wait_async() {
  asm volatile("s_wait_asynccnt 0x0" ::: "memory");
}
__device__ __forceinline__ unsigned lds_off(const void* p) {
  return (unsigned)(uintptr_t)p;   // low 32 bits of generic LDS pointer = LDS offset
}

// Problem constants
static constexpr int Bn = 8;
static constexpr int Sn = 1024;
static constexpr int En = 1024;   // E
static constexpr int On = 1024;   // O
static constexpr int Hn = 16;
static constexpr int Dn = 64;
static constexpr int SB = Sn * Bn;   // 8192 rows

// ---------------------------------------------------------------------------
// f32 -> f16 elementwise convert (weights)
// ---------------------------------------------------------------------------
__global__ __launch_bounds__(256) void cvt_f16(const float* __restrict__ x,
                                               _Float16* __restrict__ y, size_t n) {
  size_t i = (size_t)blockIdx.x * blockDim.x + threadIdx.x;
  size_t stride = (size_t)gridDim.x * blockDim.x;
  for (; i < n; i += stride) y[i] = (_Float16)x[i];
}

// (B,S,C) f32 -> (S*B, C) f16 with row = s*B + b (matches swapaxes(.,0,1))
__global__ __launch_bounds__(256) void cvt_bsq_to_sb(const float* __restrict__ x,
                                                     _Float16* __restrict__ y) {
  int row = blockIdx.x;            // 0..SB-1, row = s*B + b
  int b = row % Bn, s = row / Bn;
  const float* src = x + ((size_t)b * Sn + s) * On;
  _Float16* dst = y + (size_t)row * On;
  for (int c = threadIdx.x; c < On; c += blockDim.x) dst[c] = (_Float16)src[c];
}

// ---------------------------------------------------------------------------
// Generic GEMM: C[m,n] = sum_k A[m,k] * W[n,k]   (A: MxK f16, W: NxK f16)
// Block tile 128x64, K-step 32; 8 waves; each wave a 32x32 patch (2x2 WMMA).
// Tile staging via GLOBAL_LOAD_ASYNC_TO_LDS_B128 (ASYNCcnt).
// ---------------------------------------------------------------------------
__global__ __launch_bounds__(256) void gemm_f16(const _Float16* __restrict__ A,
                                                const _Float16* __restrict__ W,
                                                _Float16* __restrict__ C16,
                                                float* __restrict__ C32,
                                                int M, int N, int K) {
  // lA: [128][40], lW: [64][40] — single flat segment for stable LDS offsets
  __shared__ _Float16 smem[128 * 40 + 64 * 40];
  _Float16* lA = smem;
  _Float16* lW = smem + 128 * 40;

  const int tid   = threadIdx.x;
  const int lane  = tid & 31;
  const int wave  = tid >> 5;
  const int waveM = wave >> 1;       // 0..3
  const int waveN = wave & 1;        // 0..1
  const int m0 = blockIdx.x * 128;
  const int n0 = blockIdx.y * 64;

  const int lm = lane & 15;          // fragment row (A) / col (B)
  const int kb = (lane >> 4) * 8;    // K sub-offset per ISA 16-bit layout

  // Per-thread staging coordinates (constant across K loop)
  const int ca0 = tid * 2, ca1 = tid * 2 + 1;          // A chunks 0..511
  const int ra0 = ca0 >> 2, cc0 = (ca0 & 3) * 8;
  const int ra1 = ca1 >> 2, cc1 = (ca1 & 3) * 8;
  const int rw  = tid >> 2, cw = (tid & 3) * 8;        // W chunk
  const unsigned la0 = lds_off(&lA[ra0 * 40 + cc0]);
  const unsigned la1 = lds_off(&lA[ra1 * 40 + cc1]);
  const unsigned lw0 = lds_off(&lW[rw * 40 + cw]);

  v8f acc[2][2];
  v8f zero = {};
  for (int i = 0; i < 2; ++i)
    for (int j = 0; j < 2; ++j) acc[i][j] = zero;

  for (int k0 = 0; k0 < K; k0 += 32) {
    __syncthreads();
    // Async-stage A tile (128x32) and W tile (64x32): 16B per async op
    async_copy_b128(la0, &A[(size_t)(m0 + ra0) * K + k0 + cc0]);
    async_copy_b128(la1, &A[(size_t)(m0 + ra1) * K + k0 + cc1]);
    async_copy_b128(lw0, &W[(size_t)(n0 + rw) * K + k0 + cw]);
    // CDNA5 global prefetch of the next K tile (global_prefetch_b8)
    if (k0 + 32 < K) {
      __builtin_prefetch(&A[(size_t)(m0 + (tid >> 1)) * K + k0 + 32], 0, 1);
      if (tid < 64) __builtin_prefetch(&W[(size_t)(n0 + tid) * K + k0 + 32], 0, 1);
    }
    wait_async();
    __syncthreads();

    v16h af[2], bf[2];
    #pragma unroll
    for (int i = 0; i < 2; ++i) {
      const _Float16* ap = &lA[(waveM * 32 + i * 16 + lm) * 40];
      const _Float16* bp = &lW[(waveN * 32 + i * 16 + lm) * 40];
      #pragma unroll
      for (int e = 0; e < 8; ++e) {
        af[i][e]     = ap[kb + e];
        af[i][8 + e] = ap[16 + kb + e];
        bf[i][e]     = bp[kb + e];
        bf[i][8 + e] = bp[16 + kb + e];
      }
    }
    #pragma unroll
    for (int i = 0; i < 2; ++i)
      #pragma unroll
      for (int j = 0; j < 2; ++j)
        acc[i][j] = WMMA_F16(af[i], bf[j], acc[i][j]);
  }

  // Epilogue: C layout — VGPR r holds M=r (lanes 0-15) / M=r+8 (lanes 16-31)
  const int half8 = (lane >> 4) * 8;
  #pragma unroll
  for (int i = 0; i < 2; ++i)
    #pragma unroll
    for (int j = 0; j < 2; ++j)
      #pragma unroll
      for (int r = 0; r < 8; ++r) {
        int m = m0 + waveM * 32 + i * 16 + r + half8;
        int n = n0 + waveN * 32 + j * 16 + lm;
        if (C16) C16[(size_t)m * N + n] = (_Float16)acc[i][j][r];
        else     C32[(size_t)m * N + n] = acc[i][j][r];
      }
}

// ---------------------------------------------------------------------------
// RoPE (positions keyed by batch index b). In-place on (S*B, E) f16.
// One thread per (row, h, d<32).
// ---------------------------------------------------------------------------
__global__ __launch_bounds__(256) void rope_kernel(_Float16* __restrict__ q) {
  size_t idx = (size_t)blockIdx.x * blockDim.x + threadIdx.x;  // SB*H*32 total
  int d   = (int)(idx & 31);
  int h   = (int)((idx >> 5) & 15);
  int row = (int)(idx >> 9);
  if (row >= SB) return;
  int b = row % Bn;
  float inv_freq = __powf(10000.0f, -(2.0f * d) / 64.0f);
  float ang = (float)b * inv_freq;
  float sn, cs;
  __sincosf(ang, &sn, &cs);
  _Float16* p = q + (size_t)row * En + h * Dn;
  float x1 = (float)p[d];
  float x2 = (float)p[d + 32];
  p[d]      = (_Float16)(x1 * cs - x2 * sn);
  p[d + 32] = (_Float16)(x2 * cs + x1 * sn);
}

// ---------------------------------------------------------------------------
// Flash attention per (b,h): block = 128 q rows; 8 waves own 16 rows each.
// qp/kp/vp: (S*B, E) f16 with head slice at col h*64. out: same layout.
// K/V tiles staged with async global->LDS copies.
// ---------------------------------------------------------------------------
__global__ __launch_bounds__(256) void attn_kernel(const _Float16* __restrict__ qp,
                                                   const _Float16* __restrict__ kp,
                                                   const _Float16* __restrict__ vp,
                                                   _Float16* __restrict__ out) {
  // lK: [64][72], lV: [64][72], lP: [8][16][72]
  __shared__ _Float16 smem[64 * 72 * 2 + 8 * 16 * 72];
  _Float16* lK = smem;
  _Float16* lV = smem + 64 * 72;
  _Float16* lP = smem + 2 * 64 * 72;

  const int bh = blockIdx.x;       // b*H + h
  const int b  = bh / Hn;
  const int h  = bh % Hn;
  const int s0 = blockIdx.y * 128;

  const int tid  = threadIdx.x;
  const int lane = tid & 31;
  const int wave = tid >> 5;
  const int lm = lane & 15;
  const int kb = (lane >> 4) * 8;

  // Q fragments for this wave's 16 rows, scale 1/sqrt(64) folded in.
  v16h qa[2];
  {
    int srow = s0 + wave * 16 + lm;
    const _Float16* qrow = qp + ((size_t)srow * Bn + b) * En + h * Dn;
    #pragma unroll
    for (int ks = 0; ks < 2; ++ks)
      #pragma unroll
      for (int e = 0; e < 8; ++e) {
        qa[ks][e]     = (_Float16)((float)qrow[ks * 32 + kb + e] * 0.125f);
        qa[ks][8 + e] = (_Float16)((float)qrow[ks * 32 + 16 + kb + e] * 0.125f);
      }
  }

  // Per-thread staging coordinates
  const int ch0 = tid * 2, ch1 = tid * 2 + 1;     // 0..511
  const int tt0 = ch0 >> 3, dc0 = (ch0 & 7) * 8;
  const int tt1 = ch1 >> 3, dc1 = (ch1 & 7) * 8;
  const unsigned lk0 = lds_off(&lK[tt0 * 72 + dc0]);
  const unsigned lk1 = lds_off(&lK[tt1 * 72 + dc1]);
  const unsigned lv0 = lds_off(&lV[tt0 * 72 + dc0]);
  const unsigned lv1 = lds_off(&lV[tt1 * 72 + dc1]);

  float mrow[8], lrow[8];
  v8f oacc[4];
  v8f zero = {};
  #pragma unroll
  for (int r = 0; r < 8; ++r) { mrow[r] = -1e30f; lrow[r] = 0.0f; }
  #pragma unroll
  for (int j = 0; j < 4; ++j) oacc[j] = zero;

  for (int t0 = 0; t0 < Sn; t0 += 64) {
    __syncthreads();
    // Async-stage K,V tiles (64 t-rows x 64 d) into LDS
    {
      size_t s0g = ((size_t)(t0 + tt0) * Bn + b) * En + h * Dn + dc0;
      size_t s1g = ((size_t)(t0 + tt1) * Bn + b) * En + h * Dn + dc1;
      async_copy_b128(lk0, &kp[s0g]);
      async_copy_b128(lk1, &kp[s1g]);
      async_copy_b128(lv0, &vp[s0g]);
      async_copy_b128(lv1, &vp[s1g]);
    }
    wait_async();
    __syncthreads();

    // Scores: 4 tiles of 16x16 over t; K dim = 64 via 2 chained WMMAs
    v8f sc[4];
    #pragma unroll
    for (int j = 0; j < 4; ++j) {
      sc[j] = zero;
      #pragma unroll
      for (int ks = 0; ks < 2; ++ks) {
        v16h bf;
        const _Float16* kr = &lK[(j * 16 + lm) * 72 + ks * 32];
        #pragma unroll
        for (int e = 0; e < 8; ++e) { bf[e] = kr[kb + e]; bf[8 + e] = kr[16 + kb + e]; }
        sc[j] = WMMA_F16(qa[ks], bf, sc[j]);
      }
    }

    // Online softmax: rows live across 16-lane groups (shfl_xor masks 1..8)
    #pragma unroll
    for (int r = 0; r < 8; ++r) {
      float mx = fmaxf(fmaxf(sc[0][r], sc[1][r]), fmaxf(sc[2][r], sc[3][r]));
      #pragma unroll
      for (int off = 8; off >= 1; off >>= 1) mx = fmaxf(mx, __shfl_xor(mx, off, 32));
      float newm  = fmaxf(mrow[r], mx);
      float alpha = __expf(mrow[r] - newm);
      float ssum = 0.0f;
      #pragma unroll
      for (int j = 0; j < 4; ++j) {
        float p = __expf(sc[j][r] - newm);
        sc[j][r] = p;
        ssum += p;
      }
      #pragma unroll
      for (int off = 8; off >= 1; off >>= 1) ssum += __shfl_xor(ssum, off, 32);
      lrow[r] = lrow[r] * alpha + ssum;
      mrow[r] = newm;
      #pragma unroll
      for (int jd = 0; jd < 4; ++jd) oacc[jd][r] *= alpha;
    }

    // Spill P to per-wave LDS scratch (C layout -> row-major f16)
    {
      int Mr = (lane >> 4) * 8;
      #pragma unroll
      for (int r = 0; r < 8; ++r)
        #pragma unroll
        for (int j = 0; j < 4; ++j)
          lP[(wave * 16 + Mr + r) * 72 + j * 16 + lm] = (_Float16)sc[j][r];
    }
    __syncthreads();

    // O += P @ V : A = P (16x64 over t), B = V (t x d)
    #pragma unroll
    for (int ks = 0; ks < 2; ++ks) {
      v16h pf;
      const _Float16* pr = &lP[(wave * 16 + lm) * 72 + ks * 32];
      #pragma unroll
      for (int e = 0; e < 8; ++e) { pf[e] = pr[kb + e]; pf[8 + e] = pr[16 + kb + e]; }
      #pragma unroll
      for (int jd = 0; jd < 4; ++jd) {
        v16h vf;
        #pragma unroll
        for (int e = 0; e < 8; ++e) {
          vf[e]     = lV[(ks * 32 + kb + e) * 72 + jd * 16 + lm];
          vf[8 + e] = lV[(ks * 32 + 16 + kb + e) * 72 + jd * 16 + lm];
        }
        oacc[jd] = WMMA_F16(pf, vf, oacc[jd]);
      }
    }
  }

  // Normalize and write out (S*B, E) f16
  const int half8 = (lane >> 4) * 8;
  #pragma unroll
  for (int r = 0; r < 8; ++r) {
    float inv = 1.0f / lrow[r];
    int s = s0 + wave * 16 + r + half8;
    #pragma unroll
    for (int jd = 0; jd < 4; ++jd)
      out[((size_t)s * Bn + b) * En + h * Dn + jd * 16 + lm] =
          (_Float16)(oacc[jd][r] * inv);
  }
}

// ---------------------------------------------------------------------------
// x = query + proj_out; LayerNorm over last dim; output (B,S,O) f32.
// proj_out rows are in (s*B+b) order; output rows in (b*S+s) order.
// ---------------------------------------------------------------------------
__global__ __launch_bounds__(256) void residual_ln(const float* __restrict__ query,
                                                   const float* __restrict__ pout,
                                                   const float* __restrict__ gamma,
                                                   const float* __restrict__ beta,
                                                   float* __restrict__ out) {
  __shared__ float red[256];
  int row = blockIdx.x;            // (b*S + s)
  int bq = row / Sn, s = row % Sn;
  size_t qrow = (size_t)row * On;
  size_t prow = ((size_t)s * Bn + bq) * On;

  float vals[4];
  float sum = 0.0f;
  #pragma unroll
  for (int i = 0; i < 4; ++i) {
    int c = threadIdx.x + i * 256;
    vals[i] = query[qrow + c] + pout[prow + c];
    sum += vals[i];
  }
  red[threadIdx.x] = sum;
  __syncthreads();
  for (int off = 128; off > 0; off >>= 1) {
    if (threadIdx.x < off) red[threadIdx.x] += red[threadIdx.x + off];
    __syncthreads();
  }
  float mu = red[0] / (float)On;
  __syncthreads();

  float vs = 0.0f;
  #pragma unroll
  for (int i = 0; i < 4; ++i) { float d = vals[i] - mu; vs += d * d; }
  red[threadIdx.x] = vs;
  __syncthreads();
  for (int off = 128; off > 0; off >>= 1) {
    if (threadIdx.x < off) red[threadIdx.x] += red[threadIdx.x + off];
    __syncthreads();
  }
  float rstd = rsqrtf(red[0] / (float)On + 1e-5f);

  #pragma unroll
  for (int i = 0; i < 4; ++i) {
    int c = threadIdx.x + i * 256;
    out[qrow + c] = (vals[i] - mu) * rstd * gamma[c] + beta[c];
  }
}

// ---------------------------------------------------------------------------
// Host-side orchestration
// ---------------------------------------------------------------------------
extern "C" void kernel_launch(void* const* d_in, const int* in_sizes, int n_in,
                              void* d_out, int out_size, void* d_ws, size_t ws_size,
                              hipStream_t stream) {
  (void)in_sizes; (void)n_in; (void)out_size; (void)ws_size;

  const float* query      = (const float*)d_in[0];
  const float* key        = (const float*)d_in[1];
  const float* value      = (const float*)d_in[2];
  const float* Wq         = (const float*)d_in[3];
  const float* Wk         = (const float*)d_in[4];
  const float* Wv         = (const float*)d_in[5];
  const float* in_proj_w  = (const float*)d_in[6];   // (3E, E)
  const float* out_proj_w = (const float*)d_in[7];   // (E, E)
  const float* proj_w     = (const float*)d_in[8];   // (O, E)
  const float* gamma      = (const float*)d_in[9];
  const float* beta       = (const float*)d_in[10];
  float* out = (float*)d_out;

  // Workspace layout (f16 elements), ~144 MB total
  const size_t ROW = (size_t)SB * En;       // 8M elems
  _Float16* w16 = (_Float16*)d_ws;
  _Float16* bufA = w16 + 0 * ROW;
  _Float16* bufB = w16 + 1 * ROW;
  _Float16* bufC = w16 + 2 * ROW;
  _Float16* bufD = w16 + 3 * ROW;
  _Float16* bufE = w16 + 4 * ROW;
  _Float16* bufF = w16 + 5 * ROW;
  _Float16* hWq   = w16 + 6 * ROW;
  _Float16* hWk   = hWq   + (size_t)En * On;
  _Float16* hWv   = hWk   + (size_t)En * On;
  _Float16* hWin  = hWv   + (size_t)En * On;          // 3E x E
  _Float16* hWout = hWin  + (size_t)3 * En * En;
  _Float16* hWproj= hWout + (size_t)En * En;
  float* bufG = (float*)(hWproj + (size_t)On * En);   // proj output f32

  // 1) Convert weights to f16
  cvt_f16<<<2048, 256, 0, stream>>>(Wq,         hWq,   (size_t)En * On);
  cvt_f16<<<2048, 256, 0, stream>>>(Wk,         hWk,   (size_t)En * On);
  cvt_f16<<<2048, 256, 0, stream>>>(Wv,         hWv,   (size_t)En * On);
  cvt_f16<<<2048, 256, 0, stream>>>(in_proj_w,  hWin,  (size_t)3 * En * En);
  cvt_f16<<<2048, 256, 0, stream>>>(out_proj_w, hWout, (size_t)En * En);
  cvt_f16<<<2048, 256, 0, stream>>>(proj_w,     hWproj,(size_t)On * En);

  // 2) (B,S,O) -> (S*B, O) f16
  cvt_bsq_to_sb<<<SB, 256, 0, stream>>>(query, bufA);
  cvt_bsq_to_sb<<<SB, 256, 0, stream>>>(key,   bufC);
  cvt_bsq_to_sb<<<SB, 256, 0, stream>>>(value, bufE);

  dim3 ggrid(SB / 128, En / 64);   // 64 x 16 blocks

  // 3) q/k/v projections: X @ W.T
  gemm_f16<<<ggrid, 256, 0, stream>>>(bufA, hWq, bufB, nullptr, SB, En, On);
  gemm_f16<<<ggrid, 256, 0, stream>>>(bufC, hWk, bufD, nullptr, SB, En, On);
  gemm_f16<<<ggrid, 256, 0, stream>>>(bufE, hWv, bufF, nullptr, SB, En, On);

  // 4) RoPE on q and k (in place)
  const int rope_blocks = (SB * Hn * 32) / 256;   // 16384
  rope_kernel<<<rope_blocks, 256, 0, stream>>>(bufB);
  rope_kernel<<<rope_blocks, 256, 0, stream>>>(bufD);

  // 5) in_proj slices: qp / kp / vp
  gemm_f16<<<ggrid, 256, 0, stream>>>(bufB, hWin,                       bufA, nullptr, SB, En, En);
  gemm_f16<<<ggrid, 256, 0, stream>>>(bufD, hWin + (size_t)En * En,     bufC, nullptr, SB, En, En);
  gemm_f16<<<ggrid, 256, 0, stream>>>(bufF, hWin + (size_t)2 * En * En, bufE, nullptr, SB, En, En);

  // 6) Flash attention: grid = (B*H, S/128)
  dim3 agrid(Bn * Hn, Sn / 128);
  attn_kernel<<<agrid, 256, 0, stream>>>(bufA, bufC, bufE, bufB);

  // 7) out_proj, then proj (f32 output)
  gemm_f16<<<ggrid, 256, 0, stream>>>(bufB, hWout,  bufD, nullptr, SB, En, En);
  gemm_f16<<<ggrid, 256, 0, stream>>>(bufD, hWproj, nullptr, bufG, SB, On, En);

  // 8) residual + LayerNorm -> d_out (B,S,O) f32
  residual_ln<<<(size_t)Bn * Sn, 256, 0, stream>>>(query, bufG, gamma, beta, out);
}
